// GPT_26199300505862
// MI455X (gfx1250) — matrix operand order repaired and checked by reading, not compiled, not run on hardware
//
#include <hip/hip_runtime.h>
#include <hip/hip_bf16.h>
#include <math.h>

// ---------------------------------------------------------------------------
// GPT-2 (124M) forward pass for MI455X (gfx1250, wave32, WMMA).
// - Weights converted once per launch to bf16 [N,K]; activations in bf16.
// - GEMMs: double-buffered LDS filled by GLOBAL_LOAD_ASYNC_TO_LDS_B128
//   (ASYNCcnt), fragments via ds_load_b128, v_wmma_f32_16x16x32_bf16.
// - Attention: WMMA flash-attention (QK^T and P.V both on the WMMA pipe),
//   K staged async to LDS, V transposed in LDS, online softmax with wave32
//   16-lane butterflies.
// ---------------------------------------------------------------------------

#define GPT_V  50257
#define GPT_D  768
#define GPT_H  12
#define GPT_L  12
#define GPT_T  1024
#define GPT_B  2
#define GPT_HD 64
#define GPT_D3 2304
#define GPT_D4 3072
#define GPT_BT (GPT_B * GPT_T)   // 2048 rows (multiple of 128)

typedef __attribute__((ext_vector_type(16))) __bf16 v16bf;
typedef __attribute__((ext_vector_type(8)))  float  v8f;

__device__ __forceinline__ float gelu_f(float x) {
    const float c = 0.7978845608028654f;  // sqrt(2/pi)
    float x3 = x * x * x;
    return 0.5f * x * (1.0f + tanhf(c * (x + 0.044715f * x3)));
}

// CDNA5 async global->LDS copy (16 B per lane), tracked by ASYNCcnt.
// ldsOff: wave-relative LDS byte offset (low 32 bits of flat shared ptr).
#define ASYNC_CP_B128(ldsOff, gPtr)                                         \
    asm volatile("global_load_async_to_lds_b128 %0, %1, off"                \
                 :: "v"(ldsOff), "v"(gPtr) : "memory")
#define WAIT_ASYNC() asm volatile("s_wait_asynccnt 0x0" ::: "memory")

// ---------------------------------------------------------------------------
// Weight convert + transpose: W[KK][NN] f32 -> Wt[NN][KK] bf16 (one-time).
// ---------------------------------------------------------------------------
template <int KK, int NN>
__global__ __launch_bounds__(256) void wcvt_t_k(const float* __restrict__ W,
                                                __bf16* __restrict__ Wt) {
    int idx = blockIdx.x * 256 + threadIdx.x;   // over NN*KK, exact multiple
    int n   = idx / KK;
    int k   = idx % KK;
    Wt[idx] = (__bf16)W[(size_t)k * NN + n];
}

// Plain f32 -> bf16 (wte is already [V,D] = [N,K]).
__global__ __launch_bounds__(256) void cvt_k(const float* __restrict__ in,
                                             __bf16* __restrict__ out) {
    size_t i = (size_t)blockIdx.x * 256 + threadIdx.x;
    out[i] = (__bf16)in[i];
}

// ---------------------------------------------------------------------------
// Embedding: x[b,t,:] = wte[idx[b,t],:] + wpe[t,:]   (f32 residual stream)
// ---------------------------------------------------------------------------
__global__ __launch_bounds__(256) void embed_k(const int* __restrict__ idx,
                                               const float* __restrict__ wte,
                                               const float* __restrict__ wpe,
                                               float* __restrict__ x) {
    int bt  = blockIdx.x;
    int t   = bt % GPT_T;
    int tok = idx[bt];
    const float* we = wte + (size_t)tok * GPT_D;
    const float* pe = wpe + (size_t)t   * GPT_D;
    float*       xo = x   + (size_t)bt  * GPT_D;
    for (int d = threadIdx.x; d < GPT_D; d += 256)
        xo[d] = we[d] + pe[d];
}

// ---------------------------------------------------------------------------
// LayerNorm over last dim (768), bf16 output (feeds WMMA GEMMs).
// ---------------------------------------------------------------------------
__global__ __launch_bounds__(256) void ln_k(const float* __restrict__ x,
                                            const float* __restrict__ w,
                                            const float* __restrict__ b,
                                            __bf16* __restrict__ y) {
    __shared__ float s1[256];
    __shared__ float s2[256];
    int tid = threadIdx.x;
    int row = blockIdx.x;
    const float* xr = x + (size_t)row * GPT_D;
    __bf16*      yr = y + (size_t)row * GPT_D;

    float v0 = xr[tid], v1 = xr[tid + 256], v2 = xr[tid + 512];
    s1[tid] = v0 + v1 + v2;
    s2[tid] = v0 * v0 + v1 * v1 + v2 * v2;
    __syncthreads();
    for (int o = 128; o > 0; o >>= 1) {
        if (tid < o) { s1[tid] += s1[tid + o]; s2[tid] += s2[tid + o]; }
        __syncthreads();
    }
    float mu   = s1[0] * (1.0f / GPT_D);
    float var  = s2[0] * (1.0f / GPT_D) - mu * mu;
    float rstd = rsqrtf(var + 1e-5f);

    yr[tid]       = (__bf16)((v0 - mu) * rstd * w[tid]       + b[tid]);
    yr[tid + 256] = (__bf16)((v1 - mu) * rstd * w[tid + 256] + b[tid + 256]);
    yr[tid + 512] = (__bf16)((v2 - mu) * rstd * w[tid + 512] + b[tid + 512]);
}

// ---------------------------------------------------------------------------
// WMMA GEMM:  out[M,N] = act(A[M,K] @ Bt^T + bias) (+ res)
//   A: bf16 [M,K].  Bg: bf16 [N,K].  OUTMODE 0: f32 Cf, 1: bf16 Cb.
// BM=BN=128, BK=32; 8 waves x (32x64) = 2x4 WMMA per K-step; double-buffered
// LDS filled by async copies; fragment-major LDS (A: [row][k], B: [col][k]).
// ---------------------------------------------------------------------------
#define BM 128
#define BN 128
#define BK 32
#define LP 48   // LDS pitch in elements (96 bytes)

template <int N, int K, int ACT, bool RES, int OUTMODE>
__global__ __launch_bounds__(256) void gemm_wmma_k(const __bf16* __restrict__ A,
                                                   const __bf16* __restrict__ Bg,
                                                   const float* __restrict__ bias,
                                                   const float* __restrict__ res,
                                                   float* __restrict__ Cf,
                                                   __bf16* __restrict__ Cb) {
    constexpr int  NK    = K / BK;
    constexpr bool FULLN = (N % BN == 0);

    __shared__ __bf16 As[2][BM][LP];   // 2 x 12 KB
    __shared__ __bf16 Bs[2][BN][LP];   // 2 x 12 KB

    const int tid   = threadIdx.x;
    const int lane  = tid & 31;
    const int wid   = tid >> 5;
    const int wm    = wid >> 1;       // 0..3  (32 rows each)
    const int wn    = wid & 1;        // 0..1  (64 cols each)
    const int lhalf = lane >> 4;      // 0 or 1
    const int l16   = lane & 15;

    const int mBase = blockIdx.y * BM;
    const int nBase = blockIdx.x * BN;

    const __bf16* aP[2];
    unsigned aLds[2][2];
#pragma unroll
    for (int it = 0; it < 2; ++it) {
        int pos  = tid + it * 256;
        int r    = pos >> 2;
        int blk  = pos & 3;
        int dblk = (blk == 1) ? 2 : (blk == 2) ? 1 : blk;   // perm8
        aP[it]   = A + (size_t)(mBase + r) * K + blk * 8;
        aLds[0][it] = (unsigned)(uintptr_t)&As[0][r][blk * 8];
        aLds[1][it] = (unsigned)(uintptr_t)&As[1][r][blk * 8];
    }
    const __bf16* bP[2];
    unsigned bLds[2][2];
    bool bOK[2];
#pragma unroll
    for (int it = 0; it < 2; ++it) {
        int pos = tid + it * 256;
        int n   = pos >> 2;
        int ci  = pos & 3;
        int gnn = nBase + n;
        bOK[it] = FULLN || (gnn < N);
        bP[it]  = Bg + (size_t)gnn * K + ci * 8;
        bLds[0][it] = (unsigned)(uintptr_t)&Bs[0][n][ci * 8];
        bLds[1][it] = (unsigned)(uintptr_t)&Bs[1][n][ci * 8];
        if (!FULLN && !bOK[it]) {   // zero once; async loads skip these slots
            const uint4 z = make_uint4(0u, 0u, 0u, 0u);
            *(uint4*)&Bs[0][n][ci * 8] = z;
            *(uint4*)&Bs[1][n][ci * 8] = z;
        }
    }

#pragma unroll
    for (int it = 0; it < 2; ++it)
        ASYNC_CP_B128(aLds[0][it], (unsigned long long)(uintptr_t)aP[it]);
#pragma unroll
    for (int it = 0; it < 2; ++it)
        if (bOK[it])
            ASYNC_CP_B128(bLds[0][it], (unsigned long long)(uintptr_t)bP[it]);
    WAIT_ASYNC();
    __syncthreads();

    v8f acc[2][4] = {};

    for (int kt = 0; kt < NK; ++kt) {
        const int cur = kt & 1;
        if (kt + 1 < NK) {
            const int k0n = (kt + 1) * BK;
#pragma unroll
            for (int it = 0; it < 2; ++it)
                ASYNC_CP_B128(aLds[cur ^ 1][it],
                              (unsigned long long)(uintptr_t)(aP[it] + k0n));
#pragma unroll
            for (int it = 0; it < 2; ++it)
                if (bOK[it])
                    ASYNC_CP_B128(bLds[cur ^ 1][it],
                                  (unsigned long long)(uintptr_t)(bP[it] + k0n));
        }

        const __bf16* AsC = &As[cur][0][0];
        const __bf16* BsC = &Bs[cur][0][0];
        v16bf af[2];
        v16bf bf[4];
#pragma unroll
        for (int mi = 0; mi < 2; ++mi)
            af[mi] = *(const v16bf*)(AsC + (wm * 32 + mi * 16 + l16) * LP +
                                     lhalf * 16);
#pragma unroll
        for (int ni = 0; ni < 4; ++ni)
            bf[ni] = *(const v16bf*)(BsC + (wn * 64 + ni * 16 + l16) * LP +
                                     lhalf * 16);

#pragma unroll
        for (int mi = 0; mi < 2; ++mi)
#pragma unroll
            for (int ni = 0; ni < 4; ++ni)
                acc[mi][ni] = __builtin_amdgcn_wmma_f32_16x16x32_bf16(
                    false, af[mi], false, bf[ni], (short)0, acc[mi][ni],
                    false, false);

        if (kt + 1 < NK) WAIT_ASYNC();
        __syncthreads();
    }

#pragma unroll
    for (int ni = 0; ni < 4; ++ni) {
        int  gcol  = nBase + wn * 64 + ni * 16 + l16;
        bool colOK = FULLN || (gcol < N);
        float bv   = (bias != nullptr && colOK) ? bias[gcol] : 0.0f;
#pragma unroll
        for (int mi = 0; mi < 2; ++mi) {
#pragma unroll
            for (int r = 0; r < 8; ++r) {
                int grow = mBase + wm * 32 + mi * 16 + lhalf * 8 + r;
                if (colOK) {
                    float v = acc[mi][ni][r] + bv;
                    if (ACT == 1) v = gelu_f(v);
                    if (RES) v += res[(size_t)grow * N + gcol];
                    if (OUTMODE == 0) Cf[(size_t)grow * N + gcol] = v;
                    else              Cb[(size_t)grow * N + gcol] = (__bf16)v;
                }
            }
        }
    }
}

// ---------------------------------------------------------------------------
// WMMA flash attention. Block = 8 waves = 128 queries of one (b,h).
// Wave owns 16 queries; loop over 32-key tiles: K async-staged to LDS
// ([key][dim] == B-fragment layout), V transposed to Vt[dim][key].
// S = Q.K^T (4 WMMAs), online softmax (16-lane butterflies), P re-layout
// through per-wave LDS slab, O += P.V (4 WMMAs, rescaled).
// ---------------------------------------------------------------------------
__global__ __launch_bounds__(256) void fattn_k(const __bf16* __restrict__ qkv,
                                               __bf16* __restrict__ o) {
    __shared__ __bf16 Ks[32][64];        // 4 KB  (key-major, frag-ready)
    __shared__ __bf16 Vt[64][32];        // 4 KB  (dim-major, frag-ready)
    __shared__ __bf16 Ps[8][16][32];     // 8 KB  (per-wave P slab)

    const int tid   = threadIdx.x;
    const int lane  = tid & 31;
    const int wv    = tid >> 5;
    const int lhalf = lane >> 4;
    const int l16   = lane & 15;

    const int bh = blockIdx.x >> 3;             // b*H + h
    const int qt = blockIdx.x & 7;              // 128-query tile in T
    const int b  = bh / GPT_H;
    const int h  = bh % GPT_H;
    const int qB  = qt * 128;
    const int q16 = qB + wv * 16;               // wave's query base (in T)

    const __bf16* Qg = qkv + (size_t)(b * GPT_T) * GPT_D3 + h * GPT_HD;
    const __bf16* Kg = Qg + GPT_D;
    const __bf16* Vg = Qg + 2 * GPT_D;

    // Q fragments for both 32-dim halves (A-layout: two b128 per fragment)
    v16bf afq[2];
    {
        const __bf16* qrow = Qg + (size_t)(q16 + l16) * GPT_D3;
#pragma unroll
        for (int s = 0; s < 2; ++s) {
            union { uint4 u[2]; v16bf v; } pk;
            pk.u[0] = *(const uint4*)(qrow + s * 32 + lhalf * 16);
            pk.u[1] = *(const uint4*)(qrow + s * 32 + lhalf * 16 + 8);
            afq[s] = pk.v;
        }
    }

    float m[8], l[8];
#pragma unroll
    for (int e = 0; e < 8; ++e) { m[e] = -INFINITY; l[e] = 0.0f; }
    v8f oacc[4] = {};

    // staging maps
    const int skey = tid >> 3, sch = (tid & 7) * 8;        // K: 1 chunk/thread
    const unsigned ksLds = (unsigned)(uintptr_t)&Ks[skey][sch];
    const int vkey = tid & 31, vdg = (tid >> 5) * 8;       // V: 8 elems/thread

    const int ntiles = (qB + 128) / 32;
    for (int jt = 0; jt < ntiles; ++jt) {
        const int j0 = jt * 32;
        // ---- cooperative staging: K async, V transposed manually ----
        ASYNC_CP_B128(ksLds, (unsigned long long)(uintptr_t)
                      (Kg + (size_t)(j0 + skey) * GPT_D3 + sch));
        {
            union { uint4 u; __bf16 e[8]; } vv;
            vv.u = *(const uint4*)(Vg + (size_t)(j0 + vkey) * GPT_D3 + vdg);
#pragma unroll
            for (int i = 0; i < 8; ++i) Vt[vdg + i][vkey] = vv.e[i];
        }
        WAIT_ASYNC();
        __syncthreads();

        if (j0 <= q16 + 15) {
            // ---- S = Q.K^T : 2 key n-tiles x 2 dim k-steps ----
            v8f sacc[2] = {};
#pragma unroll
            for (int nt = 0; nt < 2; ++nt) {
                v16bf bk0 = *(const v16bf*)&Ks[nt * 16 + l16][lhalf * 16];
                v16bf bk1 = *(const v16bf*)&Ks[nt * 16 + l16][32 + lhalf * 16];
                sacc[nt] = __builtin_amdgcn_wmma_f32_16x16x32_bf16(
                    false, afq[0], false, bk0, (short)0, sacc[nt], false, false);
                sacc[nt] = __builtin_amdgcn_wmma_f32_16x16x32_bf16(
                    false, afq[1], false, bk1, (short)0, sacc[nt], false, false);
            }
            const float scale = 0.125f;            // 1/sqrt(64)
            float s0[8], s1[8];
#pragma unroll
            for (int e = 0; e < 8; ++e) {
                s0[e] = sacc[0][e] * scale;
                s1[e] = sacc[1][e] * scale;
            }
            if (j0 + 31 > q16) {                   // diagonal tiles: mask
#pragma unroll
                for (int e = 0; e < 8; ++e) {
                    int q = q16 + e + lhalf * 8;
                    if (j0 + l16 > q)      s0[e] = -INFINITY;
                    if (j0 + 16 + l16 > q) s1[e] = -INFINITY;
                }
            }
            // row max across the 16-lane half-wave
            float mx[8];
#pragma unroll
            for (int e = 0; e < 8; ++e) mx[e] = fmaxf(s0[e], s1[e]);
#pragma unroll
            for (int off = 8; off >= 1; off >>= 1)
#pragma unroll
                for (int e = 0; e < 8; ++e)
                    mx[e] = fmaxf(mx[e], __shfl_xor(mx[e], off, 32));

            float cr[8], rs[8];
#pragma unroll
            for (int e = 0; e < 8; ++e) {
                float mn = fmaxf(m[e], mx[e]);
                cr[e] = __expf(m[e] - mn);
                s0[e] = __expf(s0[e] - mn);
                s1[e] = __expf(s1[e] - mn);
                m[e]  = mn;
                rs[e] = s0[e] + s1[e];
            }
#pragma unroll
            for (int off = 8; off >= 1; off >>= 1)
#pragma unroll
                for (int e = 0; e < 8; ++e)
                    rs[e] += __shfl_xor(rs[e], off, 32);
#pragma unroll
            for (int e = 0; e < 8; ++e) l[e] = l[e] * cr[e] + rs[e];

            // ---- P -> per-wave LDS slab, reload as A-fragment ----
#pragma unroll
            for (int e = 0; e < 8; ++e) {
                Ps[wv][e + lhalf * 8][l16]      = (__bf16)s0[e];
                Ps[wv][e + lhalf * 8][16 + l16] = (__bf16)s1[e];
            }
            union { uint4 u[2]; v16bf v; } pk;
            pk.u[0] = *(const uint4*)&Ps[wv][l16][lhalf * 16];
            pk.u[1] = *(const uint4*)&Ps[wv][l16][lhalf * 16 + 8];
            v16bf afp = pk.v;

            // ---- O = O*cr + P.V  (4 WMMAs over 16-dim n-tiles) ----
#pragma unroll
            for (int d = 0; d < 4; ++d) {
#pragma unroll
                for (int e = 0; e < 8; ++e) oacc[d][e] *= cr[e];
                v16bf bv = *(const v16bf*)&Vt[d * 16 + l16][lhalf * 16];
                oacc[d] = __builtin_amdgcn_wmma_f32_16x16x32_bf16(
                    false, afp, false, bv, (short)0, oacc[d], false, false);
            }
        }
        __syncthreads();
    }

    // ---- write O / l ----
#pragma unroll
    for (int e = 0; e < 8; ++e) {
        float inv = 1.0f / l[e];
        int grow  = b * GPT_T + q16 + e + lhalf * 8;
        __bf16* op = o + (size_t)grow * GPT_D + h * GPT_HD;
#pragma unroll
        for (int d = 0; d < 4; ++d)
            op[d * 16 + l16] = (__bf16)(oacc[d][e] * inv);
    }
}

// ---------------------------------------------------------------------------
// Per-row log-sum-exp over vocab (for CE loss). One block per row.
// ---------------------------------------------------------------------------
__global__ __launch_bounds__(256) void lse_k(const float* __restrict__ logits,
                                             float* __restrict__ stats) {
    __shared__ float sh[256];
    int tid = threadIdx.x;
    int row = blockIdx.x;
    const float* lr = logits + (size_t)row * GPT_V;

    float mx = -INFINITY;
    for (int c = tid; c < GPT_V; c += 256) mx = fmaxf(mx, lr[c]);
    sh[tid] = mx;
    __syncthreads();
    for (int o = 128; o > 0; o >>= 1) {
        if (tid < o) sh[tid] = fmaxf(sh[tid], sh[tid + o]);
        __syncthreads();
    }
    mx = sh[0];
    __syncthreads();

    float sum = 0.0f;
    for (int c = tid; c < GPT_V; c += 256) sum += expf(lr[c] - mx);
    sh[tid] = sum;
    __syncthreads();
    for (int o = 128; o > 0; o >>= 1) {
        if (tid < o) sh[tid] += sh[tid + o];
        __syncthreads();
    }
    if (tid == 0) {
        stats[2 * row]     = mx;
        stats[2 * row + 1] = logf(sh[0]);
    }
}

// ---------------------------------------------------------------------------
// Mean NLL over valid targets. Single block.
// ---------------------------------------------------------------------------
__global__ __launch_bounds__(256) void loss_k(const float* __restrict__ logits,
                                              const float* __restrict__ stats,
                                              const int* __restrict__ tgt,
                                              float* __restrict__ out_loss) {
    __shared__ float sh[256];
    __shared__ float shc[256];
    int tid = threadIdx.x;
    float s = 0.0f, c = 0.0f;
    for (int r = tid; r < GPT_BT; r += 256) {
        int t = tgt[r];
        if (t != -1) {
            float nll = stats[2 * r] + stats[2 * r + 1] -
                        logits[(size_t)r * GPT_V + t];
            s += nll;
            c += 1.0f;
        }
    }
    sh[tid]  = s;
    shc[tid] = c;
    __syncthreads();
    for (int o = 128; o > 0; o >>= 1) {
        if (tid < o) { sh[tid] += sh[tid + o]; shc[tid] += shc[tid + o]; }
        __syncthreads();
    }
    if (tid == 0) out_loss[0] = sh[0] / fmaxf(shc[0], 1.0f);
}

// ---------------------------------------------------------------------------
// Host-side orchestration
// ---------------------------------------------------------------------------
extern "C" void kernel_launch(void* const* d_in, const int* in_sizes, int n_in,
                              void* d_out, int out_size, void* d_ws, size_t ws_size,
                              hipStream_t stream) {
    (void)in_sizes; (void)n_in; (void)out_size; (void)ws_size;

    const int*   idx     = (const int*)d_in[0];
    const int*   targets = (const int*)d_in[1];
    const float* wte     = (const float*)d_in[2];
    const float* wpe     = (const float*)d_in[3];
    const float* ln1_w   = (const float*)d_in[4];
    const float* ln1_b   = (const float*)d_in[5];
    const float* qkv_w   = (const float*)d_in[6];
    const float* qkv_b   = (const float*)d_in[7];
    const float* proj_w  = (const float*)d_in[8];
    const float* proj_b  = (const float*)d_in[9];
    const float* ln2_w   = (const float*)d_in[10];
    const float* ln2_b   = (const float*)d_in[11];
    const float* fc_w    = (const float*)d_in[12];
    const float* fc_b    = (const float*)d_in[13];
    const float* fc2_w   = (const float*)d_in[14];
    const float* fc2_b   = (const float*)d_in[15];
    const float* lnf_w   = (const float*)d_in[16];
    const float* lnf_b   = (const float*)d_in[17];

    float* logits = (float*)d_out;                          // [BT, V]
    float* loss   = logits + (size_t)GPT_BT * GPT_V;        // scalar

    // ---- workspace carve-up (~125 MB) ----
    char* p = (char*)d_ws;
    float*  x        = (float*)p;  p += (size_t)GPT_BT * GPT_D  * 4;  // residual
    float*  stats    = (float*)p;  p += (size_t)GPT_BT * 2      * 4;
    __bf16* qkv_bf   = (__bf16*)p; p += (size_t)GPT_BT * GPT_D3 * 2;
    __bf16* h_bf     = (__bf16*)p; p += (size_t)GPT_BT * GPT_D  * 2;
    __bf16* attno_bf = (__bf16*)p; p += (size_t)GPT_BT * GPT_D  * 2;
    __bf16* fcb_bf   = (__bf16*)p; p += (size_t)GPT_BT * GPT_D4 * 2;
    __bf16* wq_bf    = (__bf16*)p; p += (size_t)GPT_D3 * GPT_D  * 2;  // [3D, D]
    __bf16* wp_bf    = (__bf16*)p; p += (size_t)GPT_D  * GPT_D  * 2;  // [D, D]
    __bf16* wf_bf    = (__bf16*)p; p += (size_t)GPT_D4 * GPT_D  * 2;  // [4D, D]
    __bf16* wf2_bf   = (__bf16*)p; p += (size_t)GPT_D  * GPT_D4 * 2;  // [D, 4D]
    __bf16* wte_bf   = (__bf16*)p; p += (size_t)GPT_V  * GPT_D  * 2;  // [V, D]

    const dim3 blk(256);

    // one-time: wte -> bf16 (already [N,K] layout for the LM head)
    cvt_k<<<(GPT_V * GPT_D) / 256, blk, 0, stream>>>(wte, wte_bf);

    // embedding
    embed_k<<<GPT_BT, blk, 0, stream>>>(idx, wte, wpe, x);

    const dim3 gQKV(GPT_D3 / BN, GPT_BT / BM);
    const dim3 gPRJ(GPT_D  / BN, GPT_BT / BM);
    const dim3 gFC (GPT_D4 / BN, GPT_BT / BM);
    const dim3 gLM ((GPT_V + BN - 1) / BN, GPT_BT / BM);
    const int  gATT = GPT_B * GPT_H * (GPT_T / 128);   // 192 blocks

    for (int l = 0; l < GPT_L; ++l) {
        const float* l1w = ln1_w  + (size_t)l * GPT_D;
        const float* l1b = ln1_b  + (size_t)l * GPT_D;
        const float* qw  = qkv_w  + (size_t)l * GPT_D * GPT_D3;
        const float* qb  = qkv_b  + (size_t)l * GPT_D3;
        const float* pw  = proj_w + (size_t)l * GPT_D * GPT_D;
        const float* pb  = proj_b + (size_t)l * GPT_D;
        const float* l2w = ln2_w  + (size_t)l * GPT_D;
        const float* l2b = ln2_b  + (size_t)l * GPT_D;
        const float* fw  = fc_w   + (size_t)l * GPT_D * GPT_D4;
        const float* fb  = fc_b   + (size_t)l * GPT_D4;
        const float* f2w = fc2_w  + (size_t)l * GPT_D4 * GPT_D;
        const float* f2b = fc2_b  + (size_t)l * GPT_D;

        // per-layer weight conversion (transpose to [N,K] bf16)
        wcvt_t_k<GPT_D,  GPT_D3><<<(GPT_D3 * GPT_D) / 256, blk, 0, stream>>>(qw,  wq_bf);
        wcvt_t_k<GPT_D,  GPT_D ><<<(GPT_D  * GPT_D) / 256, blk, 0, stream>>>(pw,  wp_bf);
        wcvt_t_k<GPT_D,  GPT_D4><<<(GPT_D4 * GPT_D) / 256, blk, 0, stream>>>(fw,  wf_bf);
        wcvt_t_k<GPT_D4, GPT_D ><<<(GPT_D  * GPT_D4) / 256, blk, 0, stream>>>(f2w, wf2_bf);

        // h = LN1(x); qkv = h @ qw + qb  (bf16, feeds flash attention)
        ln_k<<<GPT_BT, blk, 0, stream>>>(x, l1w, l1b, h_bf);
        gemm_wmma_k<GPT_D3, GPT_D, 0, false, 1><<<gQKV, blk, 0, stream>>>(
            h_bf, wq_bf, qb, nullptr, nullptr, qkv_bf);

        // WMMA flash attention -> attno (bf16)
        fattn_k<<<gATT, blk, 0, stream>>>(qkv_bf, attno_bf);

        // x = x + attno @ pw + pb
        gemm_wmma_k<GPT_D, GPT_D, 0, true, 0><<<gPRJ, blk, 0, stream>>>(
            attno_bf, wp_bf, pb, x, x, nullptr);

        // h = LN2(x); fcb = gelu(h @ fw + fb) (bf16); x = x + fcb @ f2w + f2b
        ln_k<<<GPT_BT, blk, 0, stream>>>(x, l2w, l2b, h_bf);
        gemm_wmma_k<GPT_D4, GPT_D, 1, false, 1><<<gFC, blk, 0, stream>>>(
            h_bf, wf_bf, fb, nullptr, nullptr, fcb_bf);
        gemm_wmma_k<GPT_D, GPT_D4, 0, true, 0><<<gPRJ, blk, 0, stream>>>(
            fcb_bf, wf2_bf, f2b, x, x, nullptr);
    }

    // final LN + tied LM head: logits = LN_f(x) @ wte^T
    ln_k<<<GPT_BT, blk, 0, stream>>>(x, lnf_w, lnf_b, h_bf);
    gemm_wmma_k<GPT_V, GPT_D, 0, false, 0><<<gLM, blk, 0, stream>>>(
        h_bf, wte_bf, nullptr, nullptr, logits, nullptr);

    // cross-entropy
    lse_k<<<GPT_BT, blk, 0, stream>>>(logits, stats);
    loss_k<<<1, blk, 0, stream>>>(logits, stats, targets, loss);
}
